// CoreRNNFW_12421045420532
// MI455X (gfx1250) — compile-verified, matrix operand order checked by reading, not run on hardware
//
#include <hip/hip_runtime.h>

#define LAMBDA 0.95f
#define ETA    0.5f
#define EPS    1e-5f
#define DH     512
#define DG     256
#define TT     32
#define BB     32
#define NWH    (DH * DH)
#define NWG    (DH * DG)

typedef __attribute__((ext_vector_type(16))) __bf16 v16bf;
typedef __attribute__((ext_vector_type(8)))  float  v8f;

// ---------- bf16 helpers (RNE) ----------
__device__ inline __bf16 f2bf(float f) {
    unsigned u = __builtin_bit_cast(unsigned, f);
    unsigned r = u + 0x7FFFu + ((u >> 16) & 1u);
    unsigned short h = (unsigned short)(r >> 16);
    return __builtin_bit_cast(__bf16, h);
}
__device__ inline __bf16 cvt_bf(float f)  { return f2bf(f); }
__device__ inline __bf16 cvt_bf(__bf16 b) { return b; }

// Pack 16 operand elements for one lane of a 16x16x32 bf16 WMMA.
// K pattern per lane: s=0..7 -> K = kbase+laneHi+s ; s=8..15 -> K = kbase+16+laneHi+(s-8)
// Caller passes p0 = base + kbase + laneHi, p1 = base + kbase + 16 + laneHi (both 8 contiguous).
template <typename T>
__device__ inline v16bf pack16(const T* p0, const T* p1) {
    v16bf r;
#pragma unroll
    for (int i = 0; i < 8; ++i) { r[i] = cvt_bf(p0[i]); r[i + 8] = cvt_bf(p1[i]); }
    return r;
}

__device__ inline v8f wmma_bf16(v16bf a, v16bf b, v8f c) {
    // (neg_a, A, neg_b, B, c_mod, C, reuse_a, reuse_b)
    return __builtin_amdgcn_wmma_f32_16x16x32_bf16(false, a, false, b, (short)0, c, false, false);
}

// Select element i (0..7) of a v8f accumulator (constant-index cndmask chain).
__device__ inline float sel8(v8f a, int i) {
    float v = a[0];
    if (i == 1) v = a[1];
    if (i == 2) v = a[2];
    if (i == 3) v = a[3];
    if (i == 4) v = a[4];
    if (i == 5) v = a[5];
    if (i == 6) v = a[6];
    if (i == 7) v = a[7];
    return v;
}

// Block-wide LayerNorm + ReLU over DH elements (512 threads, 16 waves).
__device__ inline void ln_relu(const float* __restrict__ x, float* __restrict__ out,
                               __bf16* __restrict__ outbf,
                               const float* __restrict__ gam, const float* __restrict__ bet,
                               float* red1, float* red2, float* stat,
                               int tid, int lane, int wv) {
    float xi = x[tid];
    float s1 = xi, s2 = xi * xi;
#pragma unroll
    for (int off = 16; off > 0; off >>= 1) {
        s1 += __shfl_xor(s1, off);
        s2 += __shfl_xor(s2, off);
    }
    if (lane == 0) { red1[wv] = s1; red2[wv] = s2; }
    __syncthreads();
    if (tid < 16) {
        float a = red1[tid], c = red2[tid];
#pragma unroll
        for (int off = 8; off > 0; off >>= 1) {
            a += __shfl_xor(a, off);
            c += __shfl_xor(c, off);
        }
        if (tid == 0) {
            float mu = a * (1.0f / DH);
            stat[0] = mu;
            stat[1] = c * (1.0f / DH) - mu * mu;
        }
    }
    __syncthreads();
    float mu = stat[0];
    float rs = __frsqrt_rn(stat[1] + EPS);
    float v  = (xi - mu) * rs * gam[tid] + bet[tid];
    v = v > 0.0f ? v : 0.0f;
    out[tid]   = v;
    outbf[tid] = f2bf(v);
    __syncthreads();
}

// ---------- main recurrent body: one block == one batch row, loops all T steps ----------
template <typename WT>
__device__ void rnn_body(const float* __restrict__ z,
                         const WT* __restrict__ Wh,
                         const WT* __restrict__ Wg,
                         const float* __restrict__ bh,
                         const float* __restrict__ gam,
                         const float* __restrict__ bet,
                         float* __restrict__ out) {
    __shared__ __bf16 Hs[32 * DH];          // history rows:    Hs[s*DH + j]  (rows 0..30 used)
    __shared__ __bf16 HsT[DH * 32];         // history columns: HsT[j*32 + s]
    __shared__ float  h_s[DH], hb_s[DH], x_s[DH];
    __shared__ __bf16 hbf_s[DH], zbf_s[DG], w_s[32];
    __shared__ float  sc_s[32];
    __shared__ float  lpow[TT];
    __shared__ float  red1[16], red2[16], stat[2];

    const int tid    = threadIdx.x;        // 512 threads = 16 waves (wave32)
    const int lane   = tid & 31;
    const int wv     = tid >> 5;
    const int laneHi = (lane >= 16) ? 8 : 0;
    const int mrow   = lane & 15;
    const int b      = blockIdx.x;

    const __bf16 zbf = f2bf(0.0f);
    for (int i = tid; i < 32 * DH; i += DH) { Hs[i] = zbf; HsT[i] = zbf; }
    h_s[tid]   = 0.0f;
    hbf_s[tid] = zbf;
    if (tid < TT) lpow[tid] = __powf(LAMBDA, (float)tid);
    __syncthreads();

#pragma unroll 1
    for (int t = 0; t < TT; ++t) {
        // ---- stage z_t[b,:] ----
        const float* zp = z + ((size_t)t * BB + b) * DG;
        if (tid < DG) zbf_s[tid] = f2bf(zp[tid]);
        __syncthreads();

        // ---- phase 1: h_base = Wh @ h + Wg @ z + bh  (WMMA, N-replicated vector) ----
        // Each wave owns M-tiles {wv, wv+16}; one shared B operand per K-tile feeds both.
        {
            const WT* rowh0 = Wh + (size_t)(wv * 16 + mrow) * DH;
            const WT* rowh1 = Wh + (size_t)((wv + 16) * 16 + mrow) * DH;
            const WT* rowg0 = Wg + (size_t)(wv * 16 + mrow) * DG;
            const WT* rowg1 = Wg + (size_t)((wv + 16) * 16 + mrow) * DG;
            v8f acc0 = {}, acc1 = {};
#pragma unroll 2
            for (int kt = 0; kt < DH / 32; ++kt) {
                v16bf bb = pack16(hbf_s + kt * 32 + laneHi, hbf_s + kt * 32 + 16 + laneHi);
                v16bf a0 = pack16(rowh0 + kt * 32 + laneHi, rowh0 + kt * 32 + 16 + laneHi);
                v16bf a1 = pack16(rowh1 + kt * 32 + laneHi, rowh1 + kt * 32 + 16 + laneHi);
                acc0 = wmma_bf16(a0, bb, acc0);
                acc1 = wmma_bf16(a1, bb, acc1);
            }
#pragma unroll 2
            for (int kt = 0; kt < DG / 32; ++kt) {
                v16bf bb = pack16(zbf_s + kt * 32 + laneHi, zbf_s + kt * 32 + 16 + laneHi);
                v16bf a0 = pack16(rowg0 + kt * 32 + laneHi, rowg0 + kt * 32 + 16 + laneHi);
                v16bf a1 = pack16(rowg1 + kt * 32 + laneHi, rowg1 + kt * 32 + 16 + laneHi);
                acc0 = wmma_bf16(a0, bb, acc0);
                acc1 = wmma_bf16(a1, bb, acc1);
            }
            // D layout: lanes 0-15 hold M=vgpr, lanes 16-31 hold M=vgpr+8 (all N cols equal)
            int r = lane & 7;
            if ((lane & 8) == 0) {
                int rowoff = ((lane & 16) ? 8 : 0) + r;
                int row0 = wv * 16 + rowoff;
                int row1 = (wv + 16) * 16 + rowoff;
                hb_s[row0] = sel8(acc0, r) + bh[row0];
                hb_s[row1] = sel8(acc1, r) + bh[row1];
            }
        }
        __syncthreads();

        // ---- h = relu(LN(h_base)) ----
        ln_relu(hb_s, h_s, hbf_s, gam, bet, red1, red2, stat, tid, lane, wv);

        // ---- inner fast-weight reads via factorized A: Ah = eta * H^T (lambda-weights * (H h)) ----
        if (t > 0) {
#pragma unroll 1
            for (int it = 0; it < 2; ++it) {
                // GEMV1: scores[s] = H[s,:] . h   (waves 0,1 -> s-tiles 0,1)
                if (wv < 2) {
                    const __bf16* rh = Hs + (size_t)(wv * 16 + mrow) * DH;
                    v8f acc = {};
#pragma unroll 2
                    for (int kt = 0; kt < DH / 32; ++kt) {
                        v16bf a  = pack16(rh + kt * 32 + laneHi, rh + kt * 32 + 16 + laneHi);
                        v16bf bb = pack16(hbf_s + kt * 32 + laneHi, hbf_s + kt * 32 + 16 + laneHi);
                        acc = wmma_bf16(a, bb, acc);
                    }
                    int r = lane & 7;
                    if ((lane & 8) == 0) {
                        int row = wv * 16 + ((lane & 16) ? 8 : 0) + r;
                        sc_s[row] = sel8(acc, r);
                    }
                }
                __syncthreads();
                if (tid < 32) {
                    float c = (tid < t) ? ETA * lpow[t - 1 - tid] : 0.0f;
                    w_s[tid] = f2bf(c * sc_s[tid]);
                }
                __syncthreads();
                // GEMV2: x[j] = h_base[j] + sum_s w[s] * H[s,j]
                // A = replicated w, B = transposed history (contiguous b128 loads)
                {
                    v16bf a = pack16(w_s + laneHi, w_s + 16 + laneHi);  // same for both tiles
#pragma unroll
                    for (int tt2 = 0; tt2 < 2; ++tt2) {
                        int jt  = wv + tt2 * 16;
                        int col = jt * 16 + mrow;
                        const __bf16* cp = HsT + (size_t)col * 32;
                        v16bf bv = pack16(cp + laneHi, cp + 16 + laneHi);
                        v8f zc = {};
                        v8f acc = wmma_bf16(a, bv, zc);
                        if (lane < 16) x_s[col] = hb_s[col] + acc[0];  // D rows identical; N = col
                    }
                }
                __syncthreads();
                ln_relu(x_s, h_s, hbf_s, gam, bet, red1, red2, stat, tid, lane, wv);
            }
        }

        // ---- Hebbian write == append h_t to history, both layouts (not on last step) ----
        if (t < TT - 1) {
            __bf16 hv = hbf_s[tid];
            Hs[t * DH + tid]   = hv;
            HsT[tid * 32 + t]  = hv;
        }
        __syncthreads();
    }

    out[(size_t)b * DH + tid] = h_s[tid];
}

__global__ void __launch_bounds__(512, 1)
rnn_kernel_bf(const float* z, const __bf16* Wh, const __bf16* Wg,
              const float* bh, const float* gam, const float* bet, float* out) {
    rnn_body<__bf16>(z, Wh, Wg, bh, gam, bet, out);
}

__global__ void __launch_bounds__(512, 1)
rnn_kernel_f32(const float* z, const float* Wh, const float* Wg,
               const float* bh, const float* gam, const float* bet, float* out) {
    rnn_body<float>(z, Wh, Wg, bh, gam, bet, out);
}

__global__ void convert_w_kernel(const float* __restrict__ Wh, const float* __restrict__ Wg,
                                 __bf16* __restrict__ dst) {
    size_t i = (size_t)blockIdx.x * 256 + threadIdx.x;
    if (i < NWH) dst[i] = f2bf(Wh[i]);
    else if (i < (size_t)NWH + NWG) dst[i] = f2bf(Wg[i - NWH]);
}

extern "C" void kernel_launch(void* const* d_in, const int* in_sizes, int n_in,
                              void* d_out, int out_size, void* d_ws, size_t ws_size,
                              hipStream_t stream) {
    (void)in_sizes; (void)n_in; (void)out_size;
    const float* z   = (const float*)d_in[0];
    const float* Wh  = (const float*)d_in[1];
    const float* Wg  = (const float*)d_in[2];
    const float* bh  = (const float*)d_in[3];
    const float* gam = (const float*)d_in[4];
    const float* bet = (const float*)d_in[5];
    float* out = (float*)d_out;

    const size_t nconv = (size_t)NWH + NWG;
    if (ws_size >= nconv * sizeof(unsigned short)) {
        __bf16* wbf = (__bf16*)d_ws;
        convert_w_kernel<<<dim3((unsigned)((nconv + 255) / 256)), dim3(256), 0, stream>>>(Wh, Wg, wbf);
        rnn_kernel_bf<<<dim3(BB), dim3(DH), 0, stream>>>(z, wbf, wbf + NWH, bh, gam, bet, out);
    } else {
        rnn_kernel_f32<<<dim3(BB), dim3(DH), 0, stream>>>(z, Wh, Wg, bh, gam, bet, out);
    }
}
